// Semantic_predictor_63084479644090
// MI455X (gfx1250) — matrix operand order repaired and checked by reading, not compiled, not run on hardware
//
#include <hip/hip_runtime.h>

#define NB_CLASS 20
#define DIM      32
#define NHEAD    4
#define DKH      8
#define NLAYERS  4
#define VTOK     6
#define SEM_DIM  192
#define SEM_HALF 96
#define BS       16              // samples per workgroup
#define ROWS     (BS * VTOK)     // 96 token rows
#define NTHREADS 192             // 6 wave32s -> one 16-row M-tile per wave
#define FEATC    48

typedef __attribute__((ext_vector_type(16))) _Float16 v16h;
typedef __attribute__((ext_vector_type(8)))  _Float16 v8h;
typedef __attribute__((ext_vector_type(8)))  float    v8f;
typedef __attribute__((ext_vector_type(4)))  unsigned int v4u;
typedef __attribute__((ext_vector_type(8)))  int      v8i;
typedef __attribute__((ext_vector_type(4)))  int      v4i;

#if defined(__HIP_DEVICE_COMPILE__) && __has_builtin(__builtin_amdgcn_tensor_load_to_lds) && __has_builtin(__builtin_amdgcn_s_wait_tensorcnt)
#define USE_TDM 1
#else
#define USE_TDM 0
#endif

__device__ __forceinline__ float elu1(float x) { return x > 0.f ? x : (expf(x) - 1.f); }

// ---- WMMA fragment helpers (ISA 7.12.2, wave32) ---------------------------
__device__ __forceinline__ v16h load_a(const _Float16* base, int ld, int m0, int k0, int lane) {
  int m  = m0 + (lane & 15);
  int kb = k0 + ((lane >> 4) << 4);
  const _Float16* p = base + m * ld + kb;
  v8h lo = *(const v8h*)p;
  v8h hi = *(const v8h*)(p + 8);
  return __builtin_shufflevector(lo, hi, 0, 1, 2, 3, 4, 5, 6, 7, 8, 9, 10, 11, 12, 13, 14, 15);
}

__device__ __forceinline__ v16h load_b_t(const _Float16* baseT, int n0, int lane) {
  int n  = n0 + (lane & 15);
  int kk = (lane >> 4) << 4;
  return *(const v16h*)(baseT + n * 32 + kk);
}

__device__ __forceinline__ void store_d(float* dst, int ld, int m0, int n0, int lane, v8f d) {
  int n  = n0 + (lane & 15);
  int mb = m0 + ((lane >> 4) << 3);
#pragma unroll
  for (int j = 0; j < 8; ++j) dst[(mb + j) * ld + n] = d[j];
}

__device__ __forceinline__ v8f wmma16(v16h a, v16h b, v8f c) {
  return __builtin_amdgcn_wmma_f32_16x16x32_f16(false, a, false, b, (short)0, c, false, false);
}

__global__ __launch_bounds__(NTHREADS)
void semantic_predictor_kernel(
    const float* __restrict__ feat,  const float* __restrict__ occ,
    const float* __restrict__ tok_w, const float* __restrict__ tok_b,
    const float* __restrict__ wq,    const float* __restrict__ wk,
    const float* __restrict__ wvp,   const float* __restrict__ wo,
    const float* __restrict__ ln1_g, const float* __restrict__ ln1_b,
    const float* __restrict__ w1,    const float* __restrict__ b1,
    const float* __restrict__ w2,    const float* __restrict__ b2,
    const float* __restrict__ ln2_g, const float* __restrict__ ln2_b,
    const float* __restrict__ fc1_w, const float* __restrict__ fc1_b,
    const float* __restrict__ fc2_w, const float* __restrict__ fc2_b,
    const float* __restrict__ fc3_w, const float* __restrict__ fc3_b,
    float* __restrict__ out, int Ns) {
  // ---- carved LDS pool (86.4 KB) with phase overlays ----
  __shared__ __align__(64) unsigned char smem[86400];
  float*    Tf  = (float*)(smem);             // 96x32 fp32 token master (== 16x192 MLP view)
  float*    Xf  = (float*)(smem + 12288);     // post-LN1 activations
  float*    Qf  = (float*)(smem + 24576);     // Q / GEMM-out scratch
  float*    Kf  = (float*)(smem + 36864);
  float*    Vf  = (float*)(smem + 49152);
  float*    Msk = (float*)(smem + 61440);     // 96 masks
  _Float16* AH  = (_Float16*)(smem + 61824);  // f16 A staging
  _Float16* BH  = (_Float16*)(smem + 67968);  // second f16 staging
  _Float16* WH  = (_Float16*)(smem + 74112);  // 6x(32x32) per-layer weights, K-contiguous W^T
  float*    FeatT = (float*)(smem + 36864);   // 96x48 feat tile (overlays Kf/Vf in phase 0/1)
  float*    TW  = (float*)(smem + 24576);     // tok_w (21x32) staged (overlays Qf in phase 0/1)
  float*    TWb = (float*)(smem + 27264);     // tok_b (32)
  _Float16* WT  = (_Float16*)(smem + 24576);  // MLP 32xN weight slab (overlays Qf in MLP)

  const int tid  = threadIdx.x;
  const int lane = tid & 31;
  const int wv_  = tid >> 5;            // wave id 0..5 == M-tile id
  const int s0   = blockIdx.x * BS;

  // Warm L2 for the next tile's feat.
  if (s0 + BS < Ns) __builtin_prefetch(feat + (size_t)(s0 + BS) * VTOK * FEATC + tid * 24, 0, 1);

  // ---- Phase 0: feat tile -> LDS via Tensor Data Mover (fallback: plain loads) ----
#if USE_TDM
  if (wv_ == 0) {
    unsigned long long gaddr = (unsigned long long)(uintptr_t)feat +
                               (unsigned long long)s0 * VTOK * FEATC * 4ull;
    unsigned int laddr = (unsigned int)(uintptr_t)FeatT;       // LDS byte offset
    unsigned int rows  = (unsigned int)((Ns - s0) * VTOK);     // rows available from tile start
    v4u g0;
    g0[0] = 1u;                                                // count=1 (user descriptor)
    g0[1] = laddr;                                             // lds_addr
    g0[2] = (unsigned int)(gaddr & 0xffffffffull);             // global_addr[31:0]
    g0[3] = (unsigned int)((gaddr >> 32) & 0x01ffffffull) | 0x80000000u;  // addr[56:32] | type=2
    v8i g1;
    g1[0] = (int)(2u << 16);                                   // data_size=4B
    g1[1] = (int)(48u << 16);                                  // tensor_dim0 = 48
    g1[2] = (int)((rows & 0xffffu) << 16);                     // tensor_dim1 lo16
    g1[3] = (int)((rows >> 16) | (48u << 16));                 // tensor_dim1 hi16 | tile_dim0=48
    g1[4] = 96;                                                // tile_dim1=96, tile_dim2=0
    g1[5] = 48;                                                // tensor_dim0_stride = 48
    g1[6] = 0;
    g1[7] = 0;
    v4i gz = {0, 0, 0, 0};
#if __clang_major__ >= 23
    v8i gz8 = {0, 0, 0, 0, 0, 0, 0, 0};
    __builtin_amdgcn_tensor_load_to_lds(g0, g1, gz, gz, gz8, 0);
#else
    __builtin_amdgcn_tensor_load_to_lds(g0, g1, gz, gz, 0);
#endif
    __builtin_amdgcn_s_wait_tensorcnt(0);
  }
#else
  {
    const size_t base = (size_t)s0 * VTOK * FEATC;
    const size_t lim  = (size_t)Ns * VTOK * FEATC;
    for (int idx = tid; idx < ROWS * FEATC; idx += NTHREADS)
      FeatT[idx] = (base + idx < lim) ? feat[base + idx] : 0.f;
  }
#endif
  // Stage embedding weights into LDS (overlays Qf region).
  for (int idx = tid; idx < 21 * DIM; idx += NTHREADS) TW[idx] = tok_w[idx];
  if (tid < DIM) TWb[tid] = tok_b[tid];
  __syncthreads();

  // ---- masks (one thread per sample) ----
  if (tid < BS) {
    int s = tid;
    float vis[VTOK], a[VTOK], bb[VTOK];
    float sum0 = 0.f, sum1 = 0.f;
#pragma unroll
    for (int v = 0; v < VTOK; ++v) {
      vis[v] = FeatT[(s * VTOK + v) * FEATC + (FEATC - 1)];
      sum0 += vis[v];
    }
#pragma unroll
    for (int v = 0; v < VTOK; ++v) {
      a[v] = (sum0 == 0.f) ? 1.f : vis[v];
      int gr = (s0 + s) * VTOK + v;
      bb[v] = a[v] * ((s0 + s < Ns) ? occ[gr] : 1.f);
      sum1 += bb[v];
    }
#pragma unroll
    for (int v = 0; v < VTOK; ++v) {
      float b2v = (sum1 == 0.f) ? 1.f : bb[v];
      Msk[s * VTOK + v] = b2v * a[v];
    }
  }

  // ---- Phase 1: embedding. Thread owns half a row: 21 inputs cached in regs. ----
  {
    int r = tid >> 1, o0 = (tid & 1) << 4;
    const float* fr = FeatT + r * FEATC;
    float in[21];
    in[0] = fr[FEATC - 1];
#pragma unroll
    for (int j = 1; j < 21; ++j) in[j] = fr[24 + j - 1];
#pragma unroll
    for (int oi = 0; oi < 16; ++oi) {
      int o = o0 + oi;
      float acc = TWb[o];
#pragma unroll
      for (int j = 0; j < 21; ++j) acc += in[j] * TW[j * DIM + o];
      float t = elu1(acc);
      Tf[r * DIM + o] = t;
      AH[r * DIM + o] = (_Float16)t;
    }
  }
  __syncthreads();

  const float scale = 0.35355339059327373f;  // 1/sqrt(DK)

  for (int li = 0; li < NLAYERS; ++li) {
    // ---- stage this layer's six 32x32 weights K-contiguous (W^T) in f16 ----
    const float* wsrc[6] = {wq + li * 1024, wk + li * 1024, wvp + li * 1024,
                            wo + li * 1024, w1 + li * 1024, w2 + li * 1024};
    for (int idx = tid; idx < 1024; idx += NTHREADS) {
      int k = idx >> 5, n = idx & 31;  // global read coalesced along n
#pragma unroll
      for (int m = 0; m < 6; ++m) WH[m * 1024 + n * 32 + k] = (_Float16)wsrc[m][idx];
    }
    __syncthreads();

    // ---- Q/K/V projections: loads first, then WMMAs, then stores ----
    {
      v16h a   = load_a(AH, DIM, wv_ * 16, 0, lane);
      v16h bq0 = load_b_t(WH, 0, lane),           bq1 = load_b_t(WH, 16, lane);
      v16h bk0 = load_b_t(WH + 1024, 0, lane),    bk1 = load_b_t(WH + 1024, 16, lane);
      v16h bv0 = load_b_t(WH + 2048, 0, lane),    bv1 = load_b_t(WH + 2048, 16, lane);
      v8f z = {};
      v8f q0 = wmma16(a, bq0, z), q1 = wmma16(a, bq1, z);
      v8f k0 = wmma16(a, bk0, z), k1 = wmma16(a, bk1, z);
      v8f v0 = wmma16(a, bv0, z), v1 = wmma16(a, bv1, z);
      store_d(Qf, DIM, wv_ * 16, 0, lane, q0);  store_d(Qf, DIM, wv_ * 16, 16, lane, q1);
      store_d(Kf, DIM, wv_ * 16, 0, lane, k0);  store_d(Kf, DIM, wv_ * 16, 16, lane, k1);
      store_d(Vf, DIM, wv_ * 16, 0, lane, v0);  store_d(Vf, DIM, wv_ * 16, 16, lane, v1);
    }
    __syncthreads();

    // ---- attention (V=6 keys, DK=8): scalar VALU ----
    for (int t = tid; t < BS * NHEAD * VTOK; t += NTHREADS) {
      int s = t / (NHEAD * VTOK);
      int h = (t / VTOK) % NHEAD;
      int qi = t % VTOK;
      int qrow = s * VTOK + qi, doff = h * DKH;
      float qv[DKH];
#pragma unroll
      for (int d = 0; d < DKH; ++d) qv[d] = Qf[qrow * DIM + doff + d] * scale;
      float sc[VTOK], mx = -__builtin_inff();
#pragma unroll
      for (int k = 0; k < VTOK; ++k) {
        if (Msk[s * VTOK + k] == 0.f) { sc[k] = -__builtin_inff(); }
        else {
          float acc = 0.f;
#pragma unroll
          for (int d = 0; d < DKH; ++d) acc += qv[d] * Kf[(s * VTOK + k) * DIM + doff + d];
          sc[k] = acc;
        }
        mx = fmaxf(mx, sc[k]);
      }
      float den = 0.f;
#pragma unroll
      for (int k = 0; k < VTOK; ++k) { sc[k] = expf(sc[k] - mx); den += sc[k]; }
      float inv = 1.f / den;
#pragma unroll
      for (int d = 0; d < DKH; ++d) {
        float o = 0.f;
#pragma unroll
        for (int k = 0; k < VTOK; ++k) o += sc[k] * Vf[(s * VTOK + k) * DIM + doff + d];
        AH[qrow * DIM + doff + d] = (_Float16)(o * inv);
      }
    }
    __syncthreads();

    // ---- O projection -> scratch (Qf) ----
    {
      v16h a  = load_a(AH, DIM, wv_ * 16, 0, lane);
      v16h b0 = load_b_t(WH + 3 * 1024, 0, lane), b1 = load_b_t(WH + 3 * 1024, 16, lane);
      v8f z = {};
      v8f d0 = wmma16(a, b0, z), d1 = wmma16(a, b1, z);
      store_d(Qf, DIM, wv_ * 16, 0, lane, d0);
      store_d(Qf, DIM, wv_ * 16, 16, lane, d1);
    }
    __syncthreads();

    // ---- LN1: x = ln(o@wo + tokens) ----
    if (tid < ROWS) {
      int r = tid;
      float vrow[DIM], mean = 0.f;
#pragma unroll
      for (int o = 0; o < DIM; ++o) { vrow[o] = Qf[r * DIM + o] + Tf[r * DIM + o]; mean += vrow[o]; }
      mean *= (1.f / DIM);
      float var = 0.f;
#pragma unroll
      for (int o = 0; o < DIM; ++o) { float d = vrow[o] - mean; var += d * d; }
      float inv = rsqrtf(var * (1.f / DIM) + 1e-6f);
#pragma unroll
      for (int o = 0; o < DIM; ++o) {
        float x = (vrow[o] - mean) * inv * ln1_g[li * DIM + o] + ln1_b[li * DIM + o];
        Xf[r * DIM + o] = x;
        AH[r * DIM + o] = (_Float16)x;
      }
    }
    __syncthreads();

    // ---- FFN1: h = relu(x@w1 + b1) -> BH (f16) ----
    {
      v16h a  = load_a(AH, DIM, wv_ * 16, 0, lane);
      v16h b0 = load_b_t(WH + 4 * 1024, 0, lane), bb1 = load_b_t(WH + 4 * 1024, 16, lane);
      v8f z = {};
      v8f d0 = wmma16(a, b0, z), d1 = wmma16(a, bb1, z);
      int nl = lane & 15;
      int mb = wv_ * 16 + ((lane >> 4) << 3);
      float bias0 = b1[li * DIM + nl], bias1 = b1[li * DIM + 16 + nl];
#pragma unroll
      for (int j = 0; j < 8; ++j) {
        BH[(mb + j) * DIM + nl]      = (_Float16)fmaxf(d0[j] + bias0, 0.f);
        BH[(mb + j) * DIM + 16 + nl] = (_Float16)fmaxf(d1[j] + bias1, 0.f);
      }
    }
    __syncthreads();

    // ---- FFN2: h@w2 -> scratch (Qf) ----
    {
      v16h a  = load_a(BH, DIM, wv_ * 16, 0, lane);
      v16h b0 = load_b_t(WH + 5 * 1024, 0, lane), bb1 = load_b_t(WH + 5 * 1024, 16, lane);
      v8f z = {};
      v8f d0 = wmma16(a, b0, z), d1 = wmma16(a, bb1, z);
      store_d(Qf, DIM, wv_ * 16, 0, lane, d0);
      store_d(Qf, DIM, wv_ * 16, 16, lane, d1);
    }
    __syncthreads();

    // ---- LN2: tokens = ln(x + h@w2 + b2) ----
    if (tid < ROWS) {
      int r = tid;
      float vrow[DIM], mean = 0.f;
#pragma unroll
      for (int o = 0; o < DIM; ++o) {
        vrow[o] = Qf[r * DIM + o] + Xf[r * DIM + o] + b2[li * DIM + o];
        mean += vrow[o];
      }
      mean *= (1.f / DIM);
      float var = 0.f;
#pragma unroll
      for (int o = 0; o < DIM; ++o) { float d = vrow[o] - mean; var += d * d; }
      float inv = rsqrtf(var * (1.f / DIM) + 1e-6f);
#pragma unroll
      for (int o = 0; o < DIM; ++o) {
        float x = (vrow[o] - mean) * inv * ln2_g[li * DIM + o] + ln2_b[li * DIM + o];
        Tf[r * DIM + o] = x;
        AH[r * DIM + o] = (_Float16)x;
      }
    }
    __syncthreads();
  }

  // ---- MLP head: AH(96x32 f16) == t(16x192) row-major; K-slabs staged in WT ----
  // fc1: thread owns column n=tid; division-free staging.
  {
    int n0a = wv_ * 32, n0b = wv_ * 32 + 16;
    v8f acc0 = {}, acc1 = {};
    for (int ks = 0; ks < 6; ++ks) {
      __syncthreads();  // previous slab fully consumed
      {
        const float* src = fc1_w + (size_t)(ks * 32) * SEM_DIM + tid;
        _Float16* dst = WT + tid * 32;
#pragma unroll
        for (int k = 0; k < 32; ++k) dst[k] = (_Float16)src[k * SEM_DIM];
      }
      __syncthreads();
      v16h a = load_a(AH, SEM_DIM, 0, ks * 32, lane);
      acc0 = wmma16(a, load_b_t(WT, n0a, lane), acc0);
      acc1 = wmma16(a, load_b_t(WT, n0b, lane), acc1);
    }
    int sb = (lane >> 4) << 3;
    int na = n0a + (lane & 15), nb = n0b + (lane & 15);
    float ba = fc1_b[na], bbv = fc1_b[nb];
#pragma unroll
    for (int j = 0; j < 8; ++j) {
      BH[(j + sb) * SEM_DIM + na] = (_Float16)elu1(acc0[j] + ba);
      BH[(j + sb) * SEM_DIM + nb] = (_Float16)elu1(acc1[j] + bbv);
    }
  }
  __syncthreads();

  // fc2: [16x192]@[192x96]+b -> elu -> AH (16x96 f16). One N-tile per wave.
  {
    int n0 = wv_ * 16;
    int nn = (tid < SEM_HALF) ? tid : tid - SEM_HALF;  // column owned for staging
    int kb = (tid < SEM_HALF) ? 0 : 16;
    v8f acc = {};
    for (int ks = 0; ks < 6; ++ks) {
      __syncthreads();
      {
        const float* src = fc2_w + (size_t)(ks * 32 + kb) * SEM_HALF + nn;
        _Float16* dst = WT + nn * 32 + kb;
#pragma unroll
        for (int k = 0; k < 16; ++k) dst[k] = (_Float16)src[k * SEM_HALF];
      }
      __syncthreads();
      v16h a = load_a(BH, SEM_DIM, 0, ks * 32, lane);
      acc = wmma16(a, load_b_t(WT, n0, lane), acc);
    }
    int sb = (lane >> 4) << 3;
    int n  = n0 + (lane & 15);
    float bias = fc2_b[n];
    __syncthreads();
#pragma unroll
    for (int j = 0; j < 8; ++j) AH[(j + sb) * SEM_HALF + n] = (_Float16)elu1(acc[j] + bias);
  }
  __syncthreads();

  // fc3: [16x96]@[96x20]+b -> relu -> out. Waves 0,1 own the two N-tiles (N padded to 32).
  {
    int nn = tid & 31;            // column owned for staging (zero-pad n >= 20)
    int kg = tid >> 5;            // k-group 0..5
    v8f acc = {};
    for (int ks = 0; ks < 3; ++ks) {
      __syncthreads();
      for (int k = kg; k < 32; k += 6)
        WT[nn * 32 + k] = (nn < NB_CLASS) ? (_Float16)fc3_w[(ks * 32 + k) * NB_CLASS + nn]
                                          : (_Float16)0.f;
      __syncthreads();
      if (wv_ < 2) {
        v16h a = load_a(AH, SEM_HALF, 0, ks * 32, lane);
        acc = wmma16(a, load_b_t(WT, wv_ * 16, lane), acc);
      }
    }
    if (wv_ < 2) {
      int n = wv_ * 16 + (lane & 15);
      if (n < NB_CLASS) {
        int sb = (lane >> 4) << 3;
        float bias = fc3_b[n];
#pragma unroll
        for (int j = 0; j < 8; ++j) {
          int s = s0 + j + sb;
          if (s < Ns) out[(size_t)s * NB_CLASS + n] = fmaxf(acc[j] + bias, 0.f);
        }
      }
    }
  }
}

extern "C" void kernel_launch(void* const* d_in, const int* in_sizes, int n_in,
                              void* d_out, int out_size, void* d_ws, size_t ws_size,
                              hipStream_t stream) {
  const float* feat  = (const float*)d_in[0];
  const float* occ   = (const float*)d_in[1];
  const float* tok_w = (const float*)d_in[2];
  const float* tok_b = (const float*)d_in[3];
  const float* wq    = (const float*)d_in[4];
  const float* wk    = (const float*)d_in[5];
  const float* wv    = (const float*)d_in[6];
  const float* wo    = (const float*)d_in[7];
  const float* ln1_g = (const float*)d_in[8];
  const float* ln1_b = (const float*)d_in[9];
  const float* w1    = (const float*)d_in[10];
  const float* b1    = (const float*)d_in[11];
  const float* w2    = (const float*)d_in[12];
  const float* b2    = (const float*)d_in[13];
  const float* ln2_g = (const float*)d_in[14];
  const float* ln2_b = (const float*)d_in[15];
  const float* fc1_w = (const float*)d_in[16];
  const float* fc1_b = (const float*)d_in[17];
  const float* fc2_w = (const float*)d_in[18];
  const float* fc2_b = (const float*)d_in[19];
  const float* fc3_w = (const float*)d_in[20];
  const float* fc3_b = (const float*)d_in[21];
  float* out = (float*)d_out;

  int Ns = in_sizes[0] / (VTOK * FEATC);
  int grid = (Ns + BS - 1) / BS;
  semantic_predictor_kernel<<<grid, NTHREADS, 0, stream>>>(
      feat, occ, tok_w, tok_b, wq, wk, wv, wo, ln1_g, ln1_b,
      w1, b1, w2, b2, ln2_g, ln2_b, fc1_w, fc1_b, fc2_w, fc2_b,
      fc3_w, fc3_b, out, Ns);
}
